// SymbolicEmbeddingBlock_52441550684192
// MI455X (gfx1250) — matrix operand order repaired and checked by reading, not compiled, not run on hardware
//
#include <hip/hip_runtime.h>

typedef __attribute__((ext_vector_type(16))) __bf16 v16bf;
typedef __attribute__((ext_vector_type(8)))  __bf16 v8bf;
typedef __attribute__((ext_vector_type(8)))  float  v8f;

#define HIDDEN 128
#define TILE_M 128
#define TABLE_ELEMS (128 * HIDDEN)           // 16384 bf16
#define WSWZ_ELEMS  (12 * 8 * 32 * 16)       // 49152 bf16
#define WS_NEEDED   ((TABLE_ELEMS + WSWZ_ELEMS) * sizeof(unsigned short))

__device__ __forceinline__ unsigned short f2bf(float f) {
    unsigned int u = __float_as_uint(f);
    u += 0x7FFFu + ((u >> 16) & 1u);   // round-to-nearest-even
    return (unsigned short)(u >> 16);
}

// silu via v_exp_f32 + v_rcp_f32 (trans ops) instead of precise-div macro.
__device__ __forceinline__ float silu(float v) {
    return v * __builtin_amdgcn_rcpf(1.0f + __expf(-v));
}

// Assemble 16-bit A fragment: K {c0..c0+15} of its row.
__device__ __forceinline__ v16bf load_a(const unsigned short* rowp, int c0) {
    union { v16bf v; v8bf h[2]; } u;
    u.h[0] = *(const v8bf*)(rowp + c0);
    u.h[1] = *(const v8bf*)(rowp + c0 + 8);
    return u.v;
}

// ---------------------------------------------------------------------------
// Prep kernel: runs once per launch. Builds (a) bf16 VQ table, (b) W
// pre-swizzled into 32x16 bf16 WMMA B fragments, contiguous k order (same as A):
//   wswz[((ks*8 + nt)*32 + lane)*16 + s] = bf16(W[K][N])
//   K = ks*32 + (lane>=16 ? 16 : 0) + s,  N = nt*16 + (lane&15)
// ---------------------------------------------------------------------------
__global__ void symvq_prep_kernel(const int* __restrict__ code_idx,
                                  const float* __restrict__ codebook,
                                  const float* __restrict__ W,
                                  unsigned short* __restrict__ table_out,
                                  unsigned short* __restrict__ wswz_out)
{
    const int idx = blockIdx.x * 256 + threadIdx.x;
    if (idx < TABLE_ELEMS) {
        const int t = idx >> 7, c = idx & 127;
        const int cid = code_idx[t * 8 + (c >> 4)];
        table_out[idx] = f2bf(codebook[cid * 16 + (c & 15)]);
    }
    if (idx < WSWZ_ELEMS) {
        const int s    = idx & 15;
        const int lane = (idx >> 4) & 31;
        const int nt   = (idx >> 9) & 7;
        const int ks   = idx >> 12;
        const int K = ks * 32 + ((lane >= 16) ? 16 : 0) + s;
        const int N = nt * 16 + (lane & 15);
        wswz_out[idx] = f2bf(W[K * HIDDEN + N]);
    }
}

// ---------------------------------------------------------------------------
// Main fused kernel (fast path): bf16 table + pre-swizzled W from workspace.
// ---------------------------------------------------------------------------
__global__ __launch_bounds__(256) void symvq_main_kernel(
    const int*   __restrict__ x,
    const float* __restrict__ rbf,
    const int*   __restrict__ ei,
    const int*   __restrict__ ej,
    const float* __restrict__ W_rbf,
    const float* __restrict__ b_rbf,
    const float* __restrict__ b,
    const unsigned short* __restrict__ table_ws,  // [128*128] bf16
    const unsigned short* __restrict__ wswz_ws,   // [12*8*32*16] bf16
    float*       __restrict__ out,
    int E)
{
    __shared__ __align__(16) unsigned short table_lds[TABLE_ELEMS];
    __shared__ __align__(16) unsigned short r_lds[TILE_M * HIDDEN];
    __shared__ __align__(16) float wrbf_lds[HIDDEN * 8]; // [c][q0..5, bias, 0]
    __shared__ float b_lds[HIDDEN];
    __shared__ int   type_i_lds[TILE_M];
    __shared__ int   type_j_lds[TILE_M];

    const int tid   = threadIdx.x;
    const int ebase = blockIdx.x * TILE_M;

    const int wave     = tid >> 5;
    const int lane     = tid & 31;
    const int laneLow  = lane & 15;
    const int laneHalf = lane >> 4;
    const int ntBase   = wave * 16;

    // B fragments: one contiguous 32B vector load each, coalesced across lanes.
    v16bf bfrag[12];
    {
        const v16bf* wv = (const v16bf*)wswz_ws;
        #pragma unroll
        for (int ks = 0; ks < 12; ++ks)
            bfrag[ks] = wv[(ks * 8 + wave) * 32 + lane];
    }

    // ---- Stage 0a: copy bf16 VQ table into LDS with b128 moves ----
    {
        const uint4* src = (const uint4*)table_ws;
        uint4*       dst = (uint4*)table_lds;
        #pragma unroll
        for (int k = 0; k < 8; ++k)
            dst[tid + k * 256] = src[tid + k * 256];
    }
    // ---- Stage 0b: W_rbf transposed [c][q] (pad 8, bias in slot 6) ----
    for (int idx = tid; idx < 6 * HIDDEN; idx += 256) {
        const int q = idx >> 7, c = idx & 127;
        wrbf_lds[c * 8 + q] = W_rbf[idx];
    }
    if (tid < HIDDEN) {
        wrbf_lds[tid * 8 + 6] = b_rbf[tid];
        wrbf_lds[tid * 8 + 7] = 0.0f;
        b_lds[tid] = b[tid];
    }
    // ---- Stage 0c: edge endpoint types ----
    {
        const int t = tid & 127;
        int e = ebase + t; if (e >= E) e = E - 1;
        if (tid < 128) type_i_lds[t] = x[ei[e]];
        else           type_j_lds[t] = x[ej[e]];
    }
    __syncthreads();

    // ---- Stage 0d: r = silu(rbf @ W_rbf + b_rbf) -> bf16 LDS ----
    {
        const int el = tid >> 1;
        const int ch = (tid & 1) * 64;
        int e = ebase + el; if (e >= E) e = E - 1;
        const float* rb = rbf + (size_t)e * 6;
        const float r0 = rb[0], r1 = rb[1], r2 = rb[2];
        const float r3 = rb[3], r4 = rb[4], r5 = rb[5];
        for (int c = ch; c < ch + 64; ++c) {
            const float4 w0 = *(const float4*)&wrbf_lds[c * 8];
            const float4 w1 = *(const float4*)&wrbf_lds[c * 8 + 4];
            float acc = w1.z;  // bias
            acc = fmaf(r0, w0.x, acc); acc = fmaf(r1, w0.y, acc);
            acc = fmaf(r2, w0.z, acc); acc = fmaf(r3, w0.w, acc);
            acc = fmaf(r4, w1.x, acc); acc = fmaf(r5, w1.y, acc);
            r_lds[el * HIDDEN + c] = f2bf(silu(acc));
        }
    }
    __syncthreads();

    // ---- Stage 1: WMMA GEMM, wave w owns output columns [16w, 16w+16) ----
    const int co = laneHalf * 16;  // A-fragment K sub-offset for this lane half

    for (int rg = 0; rg < 8; ++rg) {
        const int row = rg * 16 + laneLow;
        const unsigned short* rowi = &table_lds[type_i_lds[row] * HIDDEN];
        const unsigned short* rowj = &table_lds[type_j_lds[row] * HIDDEN];
        const unsigned short* rowr = &r_lds[row * HIDDEN];

        // Software-pipelined: issue load for fragment k+1 before WMMA k so the
        // ds latency hides under the matrix op.
        v8f acc = {};
        v16bf a_cur = load_a(rowi, co);
        #pragma unroll
        for (int ks = 0; ks < 12; ++ks) {
            v16bf a_next = a_cur;
            if (ks < 11) {
                const int kn = ks + 1;
                const unsigned short* rp = (kn < 4) ? rowi
                                          : ((kn < 8) ? rowj : rowr);
                a_next = load_a(rp, ((kn & 3) * 32) + co);
            }
            acc = __builtin_amdgcn_wmma_f32_16x16x32_bf16(
                false, a_cur, false, bfrag[ks], (short)0, acc, false, false);
            a_cur = a_next;
        }

        // Epilogue: bias + silu; non-temporal stores (write-once 82MB stream).
        const int colOut = ntBase + laneLow;
        const float bias = b_lds[colOut];
        #pragma unroll
        for (int v = 0; v < 8; ++v) {
            const int m    = v + laneHalf * 8;
            const int erow = ebase + rg * 16 + m;
            if (erow < E)
                __builtin_nontemporal_store(silu(acc[v] + bias),
                                            &out[(size_t)erow * HIDDEN + colOut]);
        }
    }
}

// ---------------------------------------------------------------------------
// Fallback: fully self-contained kernel (no workspace), same algorithm.
// ---------------------------------------------------------------------------
__global__ __launch_bounds__(256) void symvq_standalone_kernel(
    const int*   __restrict__ x,
    const float* __restrict__ rbf,
    const int*   __restrict__ ei,
    const int*   __restrict__ ej,
    const int*   __restrict__ code_idx,
    const float* __restrict__ codebook,
    const float* __restrict__ W_rbf,
    const float* __restrict__ b_rbf,
    const float* __restrict__ W,
    const float* __restrict__ b,
    float*       __restrict__ out,
    int E)
{
    __shared__ __align__(16) unsigned short table_lds[TABLE_ELEMS];
    __shared__ __align__(16) unsigned short r_lds[TILE_M * HIDDEN];
    __shared__ __align__(16) float wrbf_lds[HIDDEN * 8];
    __shared__ float b_lds[HIDDEN];
    __shared__ int   type_i_lds[TILE_M];
    __shared__ int   type_j_lds[TILE_M];

    const int tid   = threadIdx.x;
    const int ebase = blockIdx.x * TILE_M;

    for (int idx = tid; idx < TABLE_ELEMS; idx += 256) {
        const int t = idx >> 7, c = idx & 127;
        const int cid = code_idx[t * 8 + (c >> 4)];
        table_lds[idx] = f2bf(codebook[cid * 16 + (c & 15)]);
    }
    for (int idx = tid; idx < 6 * HIDDEN; idx += 256) {
        const int q = idx >> 7, c = idx & 127;
        wrbf_lds[c * 8 + q] = W_rbf[idx];
    }
    if (tid < HIDDEN) {
        wrbf_lds[tid * 8 + 6] = b_rbf[tid];
        wrbf_lds[tid * 8 + 7] = 0.0f;
        b_lds[tid] = b[tid];
    }
    {
        const int t = tid & 127;
        int e = ebase + t; if (e >= E) e = E - 1;
        if (tid < 128) type_i_lds[t] = x[ei[e]];
        else           type_j_lds[t] = x[ej[e]];
    }
    __syncthreads();
    {
        const int el = tid >> 1;
        const int ch = (tid & 1) * 64;
        int e = ebase + el; if (e >= E) e = E - 1;
        const float* rb = rbf + (size_t)e * 6;
        const float r0 = rb[0], r1 = rb[1], r2 = rb[2];
        const float r3 = rb[3], r4 = rb[4], r5 = rb[5];
        for (int c = ch; c < ch + 64; ++c) {
            const float4 w0 = *(const float4*)&wrbf_lds[c * 8];
            const float4 w1 = *(const float4*)&wrbf_lds[c * 8 + 4];
            float acc = w1.z;
            acc = fmaf(r0, w0.x, acc); acc = fmaf(r1, w0.y, acc);
            acc = fmaf(r2, w0.z, acc); acc = fmaf(r3, w0.w, acc);
            acc = fmaf(r4, w1.x, acc); acc = fmaf(r5, w1.y, acc);
            r_lds[el * HIDDEN + c] = f2bf(silu(acc));
        }
    }
    __syncthreads();

    const int wave     = tid >> 5;
    const int lane     = tid & 31;
    const int laneLow  = lane & 15;
    const int laneHalf = lane >> 4;
    const int ntBase   = wave * 16;

    v16bf bfrag[12];
    #pragma unroll
    for (int ks = 0; ks < 12; ++ks) {
        const int kb = ks * 32 + laneHalf * 16;
        const float* wp = W + (size_t)kb * HIDDEN + ntBase + laneLow;
        union { v16bf v; unsigned short u[16]; } tmp;
        #pragma unroll
        for (int s = 0; s < 16; ++s) tmp.u[s] = f2bf(wp[(size_t)s * HIDDEN]);
        bfrag[ks] = tmp.v;
    }

    const int co = laneHalf * 16;
    for (int rg = 0; rg < 8; ++rg) {
        const int row = rg * 16 + laneLow;
        const unsigned short* rowi = &table_lds[type_i_lds[row] * HIDDEN];
        const unsigned short* rowj = &table_lds[type_j_lds[row] * HIDDEN];
        const unsigned short* rowr = &r_lds[row * HIDDEN];

        v8f acc = {};
        v16bf a_cur = load_a(rowi, co);
        #pragma unroll
        for (int ks = 0; ks < 12; ++ks) {
            v16bf a_next = a_cur;
            if (ks < 11) {
                const int kn = ks + 1;
                const unsigned short* rp = (kn < 4) ? rowi
                                          : ((kn < 8) ? rowj : rowr);
                a_next = load_a(rp, ((kn & 3) * 32) + co);
            }
            acc = __builtin_amdgcn_wmma_f32_16x16x32_bf16(
                false, a_cur, false, bfrag[ks], (short)0, acc, false, false);
            a_cur = a_next;
        }

        const int colOut = ntBase + laneLow;
        const float bias = b_lds[colOut];
        #pragma unroll
        for (int v = 0; v < 8; ++v) {
            const int m    = v + laneHalf * 8;
            const int erow = ebase + rg * 16 + m;
            if (erow < E)
                __builtin_nontemporal_store(silu(acc[v] + bias),
                                            &out[(size_t)erow * HIDDEN + colOut]);
        }
    }
}

extern "C" void kernel_launch(void* const* d_in, const int* in_sizes, int n_in,
                              void* d_out, int out_size, void* d_ws, size_t ws_size,
                              hipStream_t stream) {
    (void)n_in; (void)out_size;
    const int*   x        = (const int*)  d_in[0];
    const float* rbf      = (const float*)d_in[1];
    const int*   ei       = (const int*)  d_in[2];
    const int*   ej       = (const int*)  d_in[3];
    const int*   code_idx = (const int*)  d_in[4];
    const float* codebook = (const float*)d_in[5];
    const float* W_rbf    = (const float*)d_in[6];
    const float* b_rbf    = (const float*)d_in[7];
    const float* W        = (const float*)d_in[8];
    const float* b        = (const float*)d_in[9];
    float* out = (float*)d_out;

    const int E    = in_sizes[2];
    const int grid = (E + TILE_M - 1) / TILE_M;

    if (ws_size >= WS_NEEDED) {
        unsigned short* table_ws = (unsigned short*)d_ws;
        unsigned short* wswz_ws  = table_ws + TABLE_ELEMS;
        symvq_prep_kernel<<<(WSWZ_ELEMS + 255) / 256, 256, 0, stream>>>(
            code_idx, codebook, W, table_ws, wswz_ws);
        symvq_main_kernel<<<grid, 256, 0, stream>>>(
            x, rbf, ei, ej, W_rbf, b_rbf, b, table_ws, wswz_ws, out, E);
    } else {
        symvq_standalone_kernel<<<grid, 256, 0, stream>>>(
            x, rbf, ei, ej, code_idx, codebook, W_rbf, b_rbf, W, b, out, E);
    }
}
